// MoELayer_70145405878703
// MI455X (gfx1250) — matrix operand order repaired and checked by reading, not compiled, not run on hardware
//
#include <hip/hip_runtime.h>

typedef __attribute__((ext_vector_type(16))) _Float16 v16h;
typedef __attribute__((ext_vector_type(8)))  _Float16 v8h;
typedef __attribute__((ext_vector_type(8)))  float    v8f;

#define DDIM 1024
#define HDIM 1024
#define ODIM 1024
#define NEXP 8
#define MTILE 32
#define HPAD 1032   // LDS row stride (halfs): 2064B = 16B-aligned, rotates banks by 4/row

// ---------------------------------------------------------------------------
// Kernel 0: zero output + per-expert counters (graph replays don't re-zero ws)
// ---------------------------------------------------------------------------
__global__ void zero_init_kernel(float* __restrict__ out, size_t n, int* __restrict__ cnt) {
    size_t i = (size_t)blockIdx.x * blockDim.x + threadIdx.x;
    size_t stride = (size_t)gridDim.x * blockDim.x;
    for (; i < n; i += stride) out[i] = 0.0f;
    if (blockIdx.x == 0 && threadIdx.x < NEXP) cnt[threadIdx.x] = 0;
}

// ---------------------------------------------------------------------------
// Kernel 1: fp32 -> f16 convert (x)
// ---------------------------------------------------------------------------
__global__ void cvt_half_kernel(const float* __restrict__ src, _Float16* __restrict__ dst, size_t n) {
    size_t i = (size_t)blockIdx.x * blockDim.x + threadIdx.x;
    size_t stride = (size_t)gridDim.x * blockDim.x;
    for (; i < n; i += stride) dst[i] = (_Float16)src[i];
}

// ---------------------------------------------------------------------------
// Kernel 2: per-expert transpose+convert: src (E,R,C) fp32 -> dst (E,C,R) f16
// WMMA B fragments (per-lane column of B) become two contiguous 16B loads.
// ---------------------------------------------------------------------------
__global__ void transpose_cvt_kernel(const float* __restrict__ src, _Float16* __restrict__ dst) {
    __shared__ float tile[32][33];
    const int e  = blockIdx.z;
    const int c0 = blockIdx.x * 32;
    const int r0 = blockIdx.y * 32;
    const float* s = src + (size_t)e * DDIM * HDIM;
    _Float16*    d = dst + (size_t)e * DDIM * HDIM;
    for (int i = threadIdx.y; i < 32; i += 8)
        tile[i][threadIdx.x] = s[(size_t)(r0 + i) * 1024 + c0 + threadIdx.x];
    __syncthreads();
    for (int i = threadIdx.y; i < 32; i += 8)
        d[(size_t)(c0 + i) * 1024 + r0 + threadIdx.x] = (_Float16)tile[threadIdx.x][i];
}

// ---------------------------------------------------------------------------
// Kernel 3: router — one wave (32 lanes) per token. Top-2 + softmax, scatter
// (token, coef) into per-expert slot lists. Slot order is nondeterministic,
// but each token's output is order-independent => deterministic results.
// ---------------------------------------------------------------------------
__global__ __launch_bounds__(256) void router_kernel(
    const float* __restrict__ x, const float* __restrict__ Wr, const float* __restrict__ br,
    int* __restrict__ cnt, int* __restrict__ slot_tok, float* __restrict__ slot_coef, int T) {
    const int wave = threadIdx.x >> 5;
    const int lane = threadIdx.x & 31;
    const int t = blockIdx.x * 8 + wave;
    if (t >= T) return;

    const float* xr = x + (size_t)t * DDIM;
    float acc[NEXP] = {};
    for (int d = lane; d < DDIM; d += 32) {
        const float xv = xr[d];
        const float* w = Wr + d * NEXP;
        #pragma unroll
        for (int e = 0; e < NEXP; ++e) acc[e] += xv * w[e];
    }
    #pragma unroll
    for (int e = 0; e < NEXP; ++e) {
        float v = acc[e];
        #pragma unroll
        for (int off = 16; off > 0; off >>= 1) v += __shfl_down(v, off, 32);
        acc[e] = v;
    }
    if (lane == 0) {
        float lg[NEXP];
        #pragma unroll
        for (int e = 0; e < NEXP; ++e) lg[e] = acc[e] + br[e];
        int i0 = 0;
        #pragma unroll
        for (int e = 1; e < NEXP; ++e) if (lg[e] > lg[i0]) i0 = e;
        int i1 = (i0 == 0) ? 1 : 0;
        #pragma unroll
        for (int e = 0; e < NEXP; ++e) { if (e == i0 || e == i1) continue; if (lg[e] > lg[i1]) i1 = e; }
        const float e1  = __expf(lg[i1] - lg[i0]);
        const float inv = 1.0f / (1.0f + e1);
        const float c0 = inv, c1 = e1 * inv;
        int p0 = atomicAdd(&cnt[i0], 1);
        slot_tok[i0 * T + p0] = t; slot_coef[i0 * T + p0] = c0;
        int p1 = atomicAdd(&cnt[i1], 1);
        slot_tok[i1 * T + p1] = t; slot_coef[i1 * T + p1] = c1;
    }
}

// ---------------------------------------------------------------------------
// Kernel 4: grouped expert FFN with WMMA f16, 32-token tiles.
// Grid: NEXP * tilesPerExpert; 256 threads = 8 waves. Per block:
//   h[32x1024] = relu(Xg @ W1_e + b1)  -> LDS (dynamic, ~66KB)
//   y[32x1024] = h @ W2_e + b2; out[token] += coef * y (hw f32 atomics)
// Each wave owns a 32x64 output strip per pass; 2 passes cover 1024 columns.
// B fragments for the 4 N-tiles are kept in distinct registers so the
// scheduler can clause the loads and overlap them with WMMA issue.
// ---------------------------------------------------------------------------
__global__ __launch_bounds__(256, 2) void moe_ffn_kernel(
    const _Float16* __restrict__ xh,    // (T, D) f16
    const _Float16* __restrict__ W1t,   // (E, H, D) f16  (transposed)
    const _Float16* __restrict__ W2t,   // (E, O, H) f16  (transposed)
    const float* __restrict__ b1,       // (E, H)
    const float* __restrict__ b2,       // (E, O)
    const int* __restrict__ cnt,
    const int* __restrict__ slot_tok,
    const float* __restrict__ slot_coef,
    float* __restrict__ out, int T, int tpe) {

    extern __shared__ char smem[];
    _Float16* hS    = (_Float16*)smem;                                   // MTILE * HPAD halfs
    int*      tokS  = (int*)(smem + (size_t)MTILE * HPAD * sizeof(_Float16));
    float*    coefS = (float*)(tokS + MTILE);

    const int e    = blockIdx.x / tpe;
    const int tile = blockIdx.x - e * tpe;
    const int base = tile * MTILE;
    const int n    = cnt[e];
    if (base >= n) return;   // uniform across block

    if (threadIdx.x < MTILE) {
        const int idx = base + threadIdx.x;
        int tok = 0; float cf = 0.0f;            // padding rows: token 0, coef 0
        if (idx < n) { tok = slot_tok[e * T + idx]; cf = slot_coef[e * T + idx]; }
        tokS[threadIdx.x]  = tok;
        coefS[threadIdx.x] = cf;
    }
    __syncthreads();

    const int wave = threadIdx.x >> 5;
    const int lane = threadIdx.x & 31;
    const int lmod = lane & 15;     // row (A) / column (B,C,D) within 16
    const int lhi  = lane >> 4;     // half-wave selector per ISA fragment layout

    const v8f vzero = {0.f, 0.f, 0.f, 0.f, 0.f, 0.f, 0.f, 0.f};

    // A fragment row pointers (rows 0-15 and 16-31 of the token tile)
    const _Float16* arow0 = xh + (size_t)tokS[lmod] * DDIM + lhi * 8;
    const _Float16* arow1 = xh + (size_t)tokS[16 + lmod] * DDIM + lhi * 8;
    const _Float16* w1e   = W1t + (size_t)e * HDIM * DDIM;

    // ---------------- GEMM1: h = relu(Xg @ W1_e + b1) ----------------
    #pragma unroll
    for (int p = 0; p < 2; ++p) {
        const int ncol0 = p * 512 + wave * 64;   // 4 N-tiles of 16
        v8f acc[2][4];
        #pragma unroll
        for (int mt = 0; mt < 2; ++mt)
            #pragma unroll
            for (int j = 0; j < 4; ++j) acc[mt][j] = vzero;

        for (int k = 0; k < DDIM; k += 32) {
            const v8h a00 = *(const v8h*)(arow0 + k);
            const v8h a01 = *(const v8h*)(arow0 + k + 16);
            const v8h a10 = *(const v8h*)(arow1 + k);
            const v8h a11 = *(const v8h*)(arow1 + k + 16);
            const v16h a0 = __builtin_shufflevector(a00, a01, 0,1,2,3,4,5,6,7,8,9,10,11,12,13,14,15);
            const v16h a1 = __builtin_shufflevector(a10, a11, 0,1,2,3,4,5,6,7,8,9,10,11,12,13,14,15);
            v16h bfr[4];
            #pragma unroll
            for (int j = 0; j < 4; ++j)
                bfr[j] = __builtin_shufflevector(*(const v8h*)(w1e + (size_t)(ncol0 + j * 16 + lmod) * DDIM + k + lhi * 8), *(const v8h*)(w1e + (size_t)(ncol0 + j * 16 + lmod) * DDIM + k + 16 + lhi * 8), 0,1,2,3,4,5,6,7,8,9,10,11,12,13,14,15);
            #pragma unroll
            for (int j = 0; j < 4; ++j) {
                acc[0][j] = __builtin_amdgcn_wmma_f32_16x16x32_f16(
                    false, a0, false, bfr[j], (short)0, acc[0][j], false, false);
                acc[1][j] = __builtin_amdgcn_wmma_f32_16x16x32_f16(
                    false, a1, false, bfr[j], (short)0, acc[1][j], false, false);
            }
        }

        // bias + relu -> LDS h tile
        #pragma unroll
        for (int j = 0; j < 4; ++j) {
            const int col = ncol0 + j * 16 + lmod;
            const float bias = b1[e * HDIM + col];
            #pragma unroll
            for (int m = 0; m < 8; ++m) {
                float v0 = acc[0][j][m] + bias; v0 = v0 > 0.f ? v0 : 0.f;
                float v1 = acc[1][j][m] + bias; v1 = v1 > 0.f ? v1 : 0.f;
                hS[(size_t)(m + 8 * lhi) * HPAD + col]      = (_Float16)v0;
                hS[(size_t)(16 + m + 8 * lhi) * HPAD + col] = (_Float16)v1;
            }
        }
    }
    __syncthreads();

    // ---------------- GEMM2: y = h @ W2_e + b2, combine, accumulate ----------------
    const _Float16* w2e   = W2t + (size_t)e * ODIM * HDIM;
    const _Float16* hrow0 = hS + (size_t)lmod * HPAD + lhi * 8;
    const _Float16* hrow1 = hS + (size_t)(16 + lmod) * HPAD + lhi * 8;

    #pragma unroll
    for (int p = 0; p < 2; ++p) {
        const int ncol0 = p * 512 + wave * 64;
        v8f acc[2][4];
        #pragma unroll
        for (int mt = 0; mt < 2; ++mt)
            #pragma unroll
            for (int j = 0; j < 4; ++j) acc[mt][j] = vzero;

        for (int k = 0; k < HDIM; k += 32) {
            const v8h a00 = *(const v8h*)(hrow0 + k);        // ds_load_b128
            const v8h a01 = *(const v8h*)(hrow0 + k + 16);
            const v8h a10 = *(const v8h*)(hrow1 + k);
            const v8h a11 = *(const v8h*)(hrow1 + k + 16);
            const v16h a0 = __builtin_shufflevector(a00, a01, 0,1,2,3,4,5,6,7,8,9,10,11,12,13,14,15);
            const v16h a1 = __builtin_shufflevector(a10, a11, 0,1,2,3,4,5,6,7,8,9,10,11,12,13,14,15);
            v16h bfr[4];
            #pragma unroll
            for (int j = 0; j < 4; ++j)
                bfr[j] = __builtin_shufflevector(*(const v8h*)(w2e + (size_t)(ncol0 + j * 16 + lmod) * HDIM + k + lhi * 8), *(const v8h*)(w2e + (size_t)(ncol0 + j * 16 + lmod) * HDIM + k + 16 + lhi * 8), 0,1,2,3,4,5,6,7,8,9,10,11,12,13,14,15);
            #pragma unroll
            for (int j = 0; j < 4; ++j) {
                acc[0][j] = __builtin_amdgcn_wmma_f32_16x16x32_f16(
                    false, a0, false, bfr[j], (short)0, acc[0][j], false, false);
                acc[1][j] = __builtin_amdgcn_wmma_f32_16x16x32_f16(
                    false, a1, false, bfr[j], (short)0, acc[1][j], false, false);
            }
        }

        #pragma unroll
        for (int j = 0; j < 4; ++j) {
            const int col = ncol0 + j * 16 + lmod;
            const float b2v = b2[e * ODIM + col];
            #pragma unroll
            for (int m = 0; m < 8; ++m) {
                const int r0 = m + 8 * lhi;          // C/D layout: elem m -> row m+8*lhi
                const int r1 = 16 + r0;
                const float val0 = (acc[0][j][m] + b2v) * coefS[r0];
                const float val1 = (acc[1][j][m] + b2v) * coefS[r1];
                unsafeAtomicAdd(out + (size_t)tokS[r0] * ODIM + col, val0);  // global_atomic_add_f32
                unsafeAtomicAdd(out + (size_t)tokS[r1] * ODIM + col, val1);
            }
        }
    }
}

// ---------------------------------------------------------------------------
// Host launcher
// ---------------------------------------------------------------------------
extern "C" void kernel_launch(void* const* d_in, const int* in_sizes, int n_in,
                              void* d_out, int out_size, void* d_ws, size_t ws_size,
                              hipStream_t stream) {
    const float* x  = (const float*)d_in[0];
    const float* Wr = (const float*)d_in[1];
    const float* br = (const float*)d_in[2];
    const float* W1 = (const float*)d_in[3];
    const float* b1 = (const float*)d_in[4];
    const float* W2 = (const float*)d_in[5];
    const float* b2 = (const float*)d_in[6];
    float* out = (float*)d_out;

    const int T = in_sizes[0] / DDIM;   // B * N_OBJ tokens (16384)

    // Workspace layout (256B-aligned slabs)
    char* ws = (char*)d_ws;
    size_t off = 0;
    auto take = [&](size_t bytes) -> char* {
        char* p = ws + off;
        off = (off + bytes + 255) & ~(size_t)255;
        return p;
    };
    _Float16* xh        = (_Float16*)take((size_t)T * DDIM * sizeof(_Float16));           // 32 MB
    _Float16* W1t       = (_Float16*)take((size_t)NEXP * HDIM * DDIM * sizeof(_Float16)); // 16 MB
    _Float16* W2t       = (_Float16*)take((size_t)NEXP * ODIM * HDIM * sizeof(_Float16)); // 16 MB
    int*      cnt       = (int*)take(NEXP * sizeof(int));
    int*      slot_tok  = (int*)take((size_t)NEXP * T * sizeof(int));                     // 512 KB
    float*    slot_coef = (float*)take((size_t)NEXP * T * sizeof(float));                 // 512 KB

    const size_t n_out = (size_t)T * ODIM;
    const size_t n_x   = (size_t)T * DDIM;

    // 0) zero output + counters (must run each launch; ws/out not re-zeroed by harness)
    zero_init_kernel<<<2048, 256, 0, stream>>>(out, n_out, cnt);

    // 1) x -> f16
    cvt_half_kernel<<<2048, 256, 0, stream>>>(x, xh, n_x);

    // 2) W1, W2 -> f16 transposed (per expert)
    dim3 tg(HDIM / 32, DDIM / 32, NEXP);
    dim3 tb(32, 8);
    transpose_cvt_kernel<<<tg, tb, 0, stream>>>(W1, W1t);
    transpose_cvt_kernel<<<tg, tb, 0, stream>>>(W2, W2t);

    // 3) router + top-2 scatter
    router_kernel<<<(T + 7) / 8, 256, 0, stream>>>(x, Wr, br, cnt, slot_tok, slot_coef, T);

    // 4) grouped expert FFN (WMMA), 32-token tiles, dynamic LDS (~66 KB < 320 KB/WGP)
    const int tpe = (T + MTILE - 1) / MTILE;
    const size_t smem_bytes = (size_t)MTILE * HPAD * sizeof(_Float16) + MTILE * (sizeof(int) + sizeof(float));
    moe_ffn_kernel<<<NEXP * tpe, 256, smem_bytes, stream>>>(xh, W1t, W2t, b1, b2,
                                                            cnt, slot_tok, slot_coef, out, T, tpe);
}
